// GPSDepth_74122545594471
// MI455X (gfx1250) — matrix operand order repaired and checked by reading, not compiled, not run on hardware
//
#include <hip/hip_runtime.h>
#include <hip/hip_bf16.h>

#define NN      100000
#define NFEAT   256
#define NHID    128
#define NCLASS  40
#define DEG     16
#define EDGES   (NN * DEG)
#define AH      16
#define LALPHA  0.2f

typedef __attribute__((ext_vector_type(16))) __bf16 v16bf;
typedef __attribute__((ext_vector_type(8)))  __bf16 v8bf;
typedef __attribute__((ext_vector_type(8)))  float  v8f;

__device__ __forceinline__ __bf16 f2bf(float f) { return (__bf16)f; }

// ---------------------------------------------------------------------------
// WMMA GEMM, fully specialized:  C[M,NT*16] = A[M,K](bf16) @ W[NT*16,K](bf16)^T
// All shape params compile-time -> K-loop fully unrolled: loads + v_wmma only.
// W is pre-padded to NT*16 rows; NR = real column count (store guard only).
// Optional f32 output C (ld LDC) and bf16 mirror Cbf (runtime ld ldcbf).
// ---------------------------------------------------------------------------
template <int K, int NT, int NR, int LDA, int LDB, int LDC>
__global__ __launch_bounds__(256)
void wmma_gemm(const __bf16* __restrict__ A, const __bf16* __restrict__ W,
               const float* __restrict__ bias,
               float* __restrict__ C, __bf16* __restrict__ Cbf, int ldcbf, int M) {
  const int wave = (blockIdx.x * blockDim.x + threadIdx.x) >> 5;
  const int ntiles = (M >> 4) * NT;
  if (wave >= ntiles) return;            // wave-uniform: EXEC stays all-ones
  const int lane = threadIdx.x & 31;
  const int m0 = (wave / NT) << 4;
  const int n0 = (wave % NT) << 4;
  const int lm = lane & 15;
  const int hi = lane >> 4;              // lane half select (0/1)
  const int arow = m0 + lm;
  const int bn   = n0 + lm;
  // A lane layout: e in [0,16): k = kc + 16*hi + e
  const __bf16* aptr = A + (size_t)arow * LDA + hi * 16;
  // B lane layout: e in [0,16): k = kc + 16*hi + e
  const __bf16* bptr = W + (size_t)bn * LDB + hi * 16;

  v8f acc = {0.f,0.f,0.f,0.f,0.f,0.f,0.f,0.f};
  #pragma unroll
  for (int kc = 0; kc < K; kc += 32) {
    v8bf  alo = *(const v8bf*)(aptr + kc);
    v8bf  ahi = *(const v8bf*)(aptr + kc + 8);
    v16bf a = __builtin_shufflevector(alo, ahi,
                0,1,2,3,4,5,6,7,8,9,10,11,12,13,14,15);
    v16bf b = *(const v16bf*)(bptr + kc);
    acc = __builtin_amdgcn_wmma_f32_16x16x32_bf16(false, a, false, b, (short)0, acc,
                                                  false, false);
  }
  if ((NR % 16 == 0) || (bn < NR)) {
    const float bv = bias ? bias[bn] : 0.f;
    const int rbase = m0 + hi * 8;
    #pragma unroll
    for (int r = 0; r < 8; ++r) {
      float v = acc[r] + bv;
      if (C)   C[(size_t)(rbase + r) * LDC + bn] = v;
      if (Cbf) Cbf[(size_t)(rbase + r) * ldcbf + bn] = f2bf(v);
    }
  }
}

// ------------------------ conversion / padding helpers ----------------------
__global__ __launch_bounds__(256)
void conv_bf16_kernel(const float* __restrict__ in, __bf16* __restrict__ out,
                      long long n) {
  long long i = blockIdx.x * (long long)blockDim.x + threadIdx.x;
  if (i < n) out[i] = f2bf(in[i]);
}

// label [N,40] -> bf16 [N,64] zero-padded
__global__ __launch_bounds__(256)
void pad_label_kernel(const float* __restrict__ label, __bf16* __restrict__ out,
                      int n_nodes) {
  int idx = blockIdx.x * blockDim.x + threadIdx.x;
  int n = idx >> 6, j = idx & 63;
  if (n >= n_nodes) return;
  out[idx] = (j < NCLASS) ? f2bf(label[(size_t)n * NCLASS + j]) : f2bf(0.f);
}

// W [K,Nc] f32 -> out [Nc,Kpad] bf16 transposed, zero-padded in k
__global__ __launch_bounds__(256)
void transpose_conv_kernel(const float* __restrict__ W, __bf16* __restrict__ out,
                           int K, int Nc, int Kpad) {
  int idx = blockIdx.x * blockDim.x + threadIdx.x;
  if (idx >= Nc * Kpad) return;
  int n = idx / Kpad, k = idx % Kpad;
  out[idx] = (k < K) ? f2bf(W[(size_t)k * Nc + n]) : f2bf(0.f);
}

// W [rows,cols] f32 -> out [rpad,cols] bf16, zero rows appended
__global__ __launch_bounds__(256)
void pad_rows_conv_kernel(const float* __restrict__ W, __bf16* __restrict__ out,
                          int rows, int rpad, int cols) {
  int idx = blockIdx.x * blockDim.x + threadIdx.x;
  if (idx >= rpad * cols) return;
  out[idx] = (idx < rows * cols) ? f2bf(W[idx]) : f2bf(0.f);
}

// ---------------------------------------------------------------------------
// h[n] = aggr_n * (sum_e lpw[c]*deg[c]*t[c]) * deg[n] + (1-aggr_n)*t[n]
// also writes bf16 mirror into lower half of catbf (ld 256). Edges of node n
// are e in [16n,16n+16) (row = repeat(arange(N), DEG)).
// ---------------------------------------------------------------------------
__global__ __launch_bounds__(256)
void hupdate_kernel(const float* __restrict__ t, const int* __restrict__ col,
                    const float* __restrict__ degree, const float* __restrict__ lpw,
                    const float* __restrict__ aggrvec, float aggrs,
                    float* __restrict__ h, __bf16* __restrict__ catlo, int n_nodes) {
  int idx = blockIdx.x * blockDim.x + threadIdx.x;
  int n = idx >> 7, j = idx & 127;
  if (n >= n_nodes) return;
  const int* c = col + (size_t)n * DEG;
  float s = 0.f;
  #pragma unroll
  for (int e = 0; e < DEG; ++e) {
    int cc = c[e];
    s += lpw[cc] * degree[cc] * t[(size_t)cc * NHID + j];
  }
  float a = aggrvec ? aggrvec[n] : aggrs;
  float v = a * s * degree[n] + (1.f - a) * t[(size_t)n * NHID + j];
  h[(size_t)n * NHID + j] = v;
  catlo[(size_t)n * 256 + j] = f2bf(v);
}

// d2[n] = mean_e || mean_e'(m[col[e']]) - m[col[e]] ||_2  (eps inside sqrt)
__global__ __launch_bounds__(256)
void minipost_kernel(const float* __restrict__ m, const int* __restrict__ col,
                     float* __restrict__ d2, int n_nodes) {
  int n = blockIdx.x * blockDim.x + threadIdx.x;
  if (n >= n_nodes) return;
  const int* c = col + (size_t)n * DEG;
  float avg[AH];
  #pragma unroll
  for (int j = 0; j < AH; ++j) avg[j] = 0.f;
  for (int e = 0; e < DEG; ++e) {
    const float* mr = m + (size_t)c[e] * AH;
    #pragma unroll
    for (int j = 0; j < AH; ++j) avg[j] += mr[j];
  }
  #pragma unroll
  for (int j = 0; j < AH; ++j) avg[j] *= (1.f / DEG);
  float dsum = 0.f;
  for (int e = 0; e < DEG; ++e) {
    const float* mr = m + (size_t)c[e] * AH;
    float q = 0.f;
    #pragma unroll
    for (int j = 0; j < AH; ++j) { float d = avg[j] - mr[j]; q += d * d; }
    dsum += sqrtf(q + 1e-12f);
  }
  d2[n] = dsum * (1.f / DEG);
}

__device__ __forceinline__ float factor_mlp(const float* hs, const float* hd,
                                            const float* hdf,
                                            const float* w1, const float* b1,
                                            const float* w2, float b2) {
  float acc = b2;
  #pragma unroll 4
  for (int i = 0; i < AH; ++i) {
    const float* wr = w1 + i * (3 * (AH + 1));
    float t = b1[i];
    #pragma unroll
    for (int j = 0; j < AH + 1; ++j) t += wr[j] * hs[j];
    #pragma unroll
    for (int j = 0; j < AH + 1; ++j) t += wr[AH + 1 + j] * hd[j];
    #pragma unroll
    for (int j = 0; j < AH + 1; ++j) t += wr[2 * (AH + 1) + j] * hdf[j];
    acc += w2[i] * tanhf(t);
  }
  return 1.f / (1.f + expf(-acc));
}

// per-edge factor: f0 from (m0,d20); f1 from (m1+ab, d21+ab) reusing hdiff of m0
__global__ __launch_bounds__(256)
void factor_kernel(const float* __restrict__ m0, const float* __restrict__ d20,
                   const float* __restrict__ m1, const float* __restrict__ d21,
                   const int* __restrict__ col,
                   const float* __restrict__ lf1w, const float* __restrict__ lf1b,
                   const float* __restrict__ lf2w, const float* __restrict__ lf2b,
                   const float* __restrict__ ab,
                   float* __restrict__ f0, float* __restrict__ f1, int n_edges) {
  __shared__ float s_w1[AH * 3 * (AH + 1)];
  __shared__ float s_b1[AH], s_w2[AH], s_ab[AH + 1], s_b2;
  for (int i = threadIdx.x; i < AH * 3 * (AH + 1); i += blockDim.x) s_w1[i] = lf1w[i];
  if (threadIdx.x < AH) { s_b1[threadIdx.x] = lf1b[threadIdx.x];
                          s_w2[threadIdx.x] = lf2w[threadIdx.x]; }
  if (threadIdx.x < AH + 1) s_ab[threadIdx.x] = ab[threadIdx.x];
  if (threadIdx.x == 0) s_b2 = lf2b[0];
  __syncthreads();

  int e = blockIdx.x * blockDim.x + threadIdx.x;
  if (e >= n_edges) return;
  int r = e >> 4;          // row node (edges = repeat(arange(N), DEG))
  int c = col[e];
  float hs[AH + 1], hd[AH + 1], hdf[AH + 1];
  #pragma unroll
  for (int j = 0; j < AH; ++j) { hs[j] = m0[(size_t)r * AH + j];
                                 hd[j] = m0[(size_t)c * AH + j]; }
  hs[AH] = d20[r]; hd[AH] = d20[c];
  #pragma unroll
  for (int j = 0; j < AH + 1; ++j) hdf[j] = fabsf(hd[j] - hs[j]);
  f0[e] = factor_mlp(hs, hd, hdf, s_w1, s_b1, s_w2, s_b2);
  #pragma unroll
  for (int j = 0; j < AH; ++j) { hs[j] = m1[(size_t)r * AH + j] + s_ab[j];
                                 hd[j] = m1[(size_t)c * AH + j] + s_ab[j]; }
  hs[AH] = d21[r] + s_ab[AH]; hd[AH] = d21[c] + s_ab[AH];
  f1[e] = factor_mlp(hs, hd, hdf, s_w1, s_b1, s_w2, s_b2);
}

__global__ __launch_bounds__(256)
void rowmean_kernel(const float* __restrict__ f, float* __restrict__ out, int n_nodes) {
  int n = blockIdx.x * blockDim.x + threadIdx.x;
  if (n >= n_nodes) return;
  float s = 0.f;
  #pragma unroll
  for (int e = 0; e < DEG; ++e) s += f[(size_t)n * DEG + e];
  out[n] = s * (1.f / DEG);
}

__global__ __launch_bounds__(256)
void f2hop_kernel(const float* __restrict__ f0, const float* __restrict__ f1hop,
                  const int* __restrict__ col, float* __restrict__ out, int n_nodes) {
  int n = blockIdx.x * blockDim.x + threadIdx.x;
  if (n >= n_nodes) return;
  float s = 0.f;
  #pragma unroll
  for (int e = 0; e < DEG; ++e)
    s += f0[(size_t)n * DEG + e] * f1hop[col[(size_t)n * DEG + e]];
  out[n] = s * (1.f / DEG);
}

// leaky-relu in place on h + refresh bf16 mirror (lower half of catbf)
__global__ __launch_bounds__(256)
void leakyconv_kernel(float* __restrict__ h, __bf16* __restrict__ catlo, int n_nodes) {
  int idx = blockIdx.x * blockDim.x + threadIdx.x;
  int n = idx >> 7, j = idx & 127;
  if (n >= n_nodes) return;
  float v = h[idx];
  v = v > 0.f ? v : LALPHA * v;
  h[idx] = v;
  catlo[(size_t)n * 256 + j] = f2bf(v);
}

__global__ __launch_bounds__(256)
void logsoftmax_kernel(float* __restrict__ out, int n_nodes) {
  int n = blockIdx.x * blockDim.x + threadIdx.x;
  if (n >= n_nodes) return;
  float* r = out + (size_t)n * NCLASS;
  float mx = r[0];
  for (int j = 1; j < NCLASS; ++j) mx = fmaxf(mx, r[j]);
  float s = 0.f;
  for (int j = 0; j < NCLASS; ++j) s += expf(r[j] - mx);
  float lse = mx + logf(s);
  for (int j = 0; j < NCLASS; ++j) r[j] -= lse;
}

// ---------------------------------------------------------------------------
extern "C" void kernel_launch(void* const* d_in, const int* in_sizes, int n_in,
                              void* d_out, int out_size, void* d_ws, size_t ws_size,
                              hipStream_t stream) {
  (void)in_sizes; (void)n_in; (void)out_size; (void)ws_size;
  const float* x      = (const float*)d_in[0];
  const int*   edges  = (const int*)  d_in[1];
  const float* degree = (const float*)d_in[2];
  const float* label  = (const float*)d_in[4];
  const float* lb_w   = (const float*)d_in[5];
  const float* lb_b   = (const float*)d_in[6];
  const float* L2F    = (const float*)d_in[7];
  const float* Ws     = (const float*)d_in[8];
  const float* w2m    = (const float*)d_in[9];
  const float* b2m    = (const float*)d_in[10];
  const float* ab     = (const float*)d_in[11];
  const float* lf1w   = (const float*)d_in[12];
  const float* lf1b   = (const float*)d_in[13];
  const float* lf2w   = (const float*)d_in[14];
  const float* lf2b   = (const float*)d_in[15];
  const float* LPW    = (const float*)d_in[16];
  const float* la2w   = (const float*)d_in[17];
  const float* la2b   = (const float*)d_in[18];
  const int*   col    = edges + EDGES;           // edges[1]
  float* out = (float*)d_out;

  char* p = (char*)d_ws;
  auto alloc = [&](size_t bytes) {
    char* r = p; p += (bytes + 255) & ~(size_t)255; return r;
  };
  float*  hA     = (float*) alloc((size_t)NN * NHID * 4);
  float*  hB     = (float*) alloc((size_t)NN * NHID * 4);
  float*  m0b    = (float*) alloc((size_t)NN * AH * 4);
  float*  m1b    = (float*) alloc((size_t)NN * AH * 4);
  float*  d20    = (float*) alloc((size_t)NN * 4);
  float*  d21    = (float*) alloc((size_t)NN * 4);
  float*  f0b    = (float*) alloc((size_t)EDGES * 4);
  float*  f1b    = (float*) alloc((size_t)EDGES * 4);
  float*  f1hop  = (float*) alloc((size_t)NN * 4);
  float*  aggrb  = (float*) alloc((size_t)NN * 4);
  __bf16* catbf  = (__bf16*)alloc((size_t)NN * 256 * 2);   // [h | pre_label] bf16
  __bf16* lbw_bf = (__bf16*)alloc((size_t)NHID * NFEAT * 2);
  __bf16* l2ft_bf= (__bf16*)alloc((size_t)NHID * 64 * 2);  // L2F^T, K padded 40->64
  __bf16* wst_bf = (__bf16*)alloc((size_t)2 * NHID * 256 * 2); // Ws^T per layer
  __bf16* w2m_bf = (__bf16*)alloc((size_t)2 * AH * NHID * 2);
  __bf16* la2_bf = (__bf16*)alloc((size_t)48 * NHID * 2);  // rows padded 40->48
  // aliases (disjoint live ranges; labbf spans f0b and f1b):
  __bf16* xbf    = (__bf16*)hB;   // x bf16 [N,256]; dead before hB first written
  __bf16* labbf  = (__bf16*)f0b;  // label bf16 [N,64]; dead before factor writes f0/f1

  const int T = 256;
  auto cdiv = [](long long a, long long b) { return (int)((a + b - 1) / b); };
  auto gblk = [&](int M, int tcols) { return cdiv((long long)(M / 16) * tcols, 8); };

  // ---- one-time weight conversion (bf16, transposed/padded where needed)
  conv_bf16_kernel<<<cdiv(NHID * NFEAT, T), T, 0, stream>>>(lb_w, lbw_bf,
                                                            NHID * NFEAT);
  transpose_conv_kernel<<<cdiv(NHID * 64, T), T, 0, stream>>>(L2F, l2ft_bf,
                                                              NCLASS, NHID, 64);
  transpose_conv_kernel<<<cdiv(NHID * 256, T), T, 0, stream>>>(Ws, wst_bf,
                                                               256, NHID, 256);
  transpose_conv_kernel<<<cdiv(NHID * 256, T), T, 0, stream>>>(
      Ws + 256 * NHID, wst_bf + (size_t)NHID * 256, 256, NHID, 256);
  conv_bf16_kernel<<<cdiv(2 * AH * NHID, T), T, 0, stream>>>(w2m, w2m_bf,
                                                             2 * AH * NHID);
  pad_rows_conv_kernel<<<cdiv(48 * NHID, T), T, 0, stream>>>(la2w, la2_bf,
                                                             NCLASS, 48, NHID);
  // ---- activation conversion
  conv_bf16_kernel<<<cdiv((long long)NN * NFEAT, T), T, 0, stream>>>(
      x, xbf, (long long)NN * NFEAT);
  pad_label_kernel<<<cdiv((long long)NN * 64, T), T, 0, stream>>>(label, labbf, NN);

  // h0 = x @ lb_w.T + lb_b       (f32 + bf16 mirror into catbf[:, :128])
  wmma_gemm<256, 8, 128, 256, 256, 128><<<gblk(NN, 8), T, 0, stream>>>(
      xbf, lbw_bf, lb_b, hA, catbf, 256, NN);
  // pre_label = label @ L2F      (bf16 only, into catbf[:, 128:256])
  wmma_gemm<64, 8, 128, 64, 64, 128><<<gblk(NN, 8), T, 0, stream>>>(
      labbf, l2ft_bf, nullptr, nullptr, catbf + 128, 256, NN);

  for (int l = 0; l < 2; ++l) {
    const bool last = (l == 1);
    // t = concat([h, pre_label]) @ Ws[l]
    wmma_gemm<256, 8, 128, 256, 256, 128><<<gblk(NN, 8), T, 0, stream>>>(
        catbf, wst_bf + (size_t)l * NHID * 256, nullptr, hB, nullptr, 0, NN);
    if (!last) {  // m1 = mini(x_layer) before catbf lower half is overwritten
      wmma_gemm<128, 1, 16, 256, 128, 16><<<gblk(NN, 1), T, 0, stream>>>(
          catbf, w2m_bf + (size_t)l * AH * NHID, b2m + l * AH, m1b, nullptr, 0, NN);
    }
    // h = aggr*agg(t)*deg + (1-aggr)*t   (+ bf16 mirror, pre-leaky)
    hupdate_kernel<<<cdiv((long long)NN * NHID, T), T, 0, stream>>>(
        hB, col, degree, LPW + (size_t)l * NN, l == 0 ? nullptr : aggrb, 0.5f,
        hA, catbf, NN);
    if (!last) {  // layer-1 aggr output is dead -> skip factor pipeline there
      wmma_gemm<128, 1, 16, 256, 128, 16><<<gblk(NN, 1), T, 0, stream>>>(
          catbf, w2m_bf + (size_t)l * AH * NHID, b2m + l * AH, m0b, nullptr, 0, NN);
      minipost_kernel<<<cdiv(NN, T), T, 0, stream>>>(m0b, col, d20, NN);
      minipost_kernel<<<cdiv(NN, T), T, 0, stream>>>(m1b, col, d21, NN);
      factor_kernel<<<cdiv(EDGES, T), T, 0, stream>>>(m0b, d20, m1b, d21, col,
          lf1w + (size_t)l * AH * 3 * (AH + 1), lf1b + l * AH,
          lf2w + l * AH, lf2b + l, ab + (size_t)l * (AH + 1), f0b, f1b, EDGES);
      rowmean_kernel<<<cdiv(NN, T), T, 0, stream>>>(f1b, f1hop, NN);
      f2hop_kernel<<<cdiv(NN, T), T, 0, stream>>>(f0b, f1hop, col, aggrb, NN);
    }
    leakyconv_kernel<<<cdiv((long long)NN * NHID, T), T, 0, stream>>>(hA, catbf, NN);
  }

  // logits = h @ la2w.T + la2b   (rows of W padded 40->48; store guard bn<40)
  wmma_gemm<128, 3, 40, 256, 128, 40><<<gblk(NN, 3), T, 0, stream>>>(
      catbf, la2_bf, la2b, out, nullptr, 0, NN);
  logsoftmax_kernel<<<cdiv(NN, T), T, 0, stream>>>(out, NN);
}
